// GroupATTBLK_12927851561325
// MI455X (gfx1250) — matrix operand order repaired and checked by reading, not compiled, not run on hardware
//
#include <hip/hip_runtime.h>
#include <hip/hip_bf16.h>

// Shapes from the reference
#define B_   4
#define C_   64
#define T_   512
#define F_   160
#define D_   4
#define N_   (B_*T_)      // 2048 tokens (b*T+t)
#define EPSV 1e-5f

typedef __attribute__((ext_vector_type(16))) _Float16 v16h;
typedef __attribute__((ext_vector_type(8)))  float    v8f;

// ---------------------------------------------------------------------------
// Pass 1: AvgPool over F (F'==1). One wave per (b,c,t) row of 160*4=640 floats.
// float4 loads -> 512B contiguous per wave per iteration (coalesced).
// Writes pooled means as f16 in [D][C][N] layout for the WMMA B-matrix.
// ---------------------------------------------------------------------------
__global__ void pool_kernel(const float* __restrict__ x, _Float16* __restrict__ p16) {
    int wid  = (blockIdx.x * blockDim.x + threadIdx.x) >> 5;   // row = (b*C+c)*T+t
    int lane = threadIdx.x & 31;
    int t = wid % T_;
    int c = (wid / T_) % C_;
    int b = wid / (C_ * T_);
    const float4* xr = (const float4*)(x + (size_t)wid * (F_ * D_));
    float4 acc = make_float4(0.f, 0.f, 0.f, 0.f);
#pragma unroll
    for (int i = 0; i < F_ / 32; ++i) {               // 5 iterations
        float4 v = xr[lane + 32 * i];                  // one float4 = all 4 d for one f
        acc.x += v.x; acc.y += v.y; acc.z += v.z; acc.w += v.w;
    }
#pragma unroll
    for (int off = 16; off > 0; off >>= 1) {           // wave32 butterfly reduction
        acc.x += __shfl_xor(acc.x, off);
        acc.y += __shfl_xor(acc.y, off);
        acc.z += __shfl_xor(acc.z, off);
        acc.w += __shfl_xor(acc.w, off);
    }
    if (lane < 4) {                                    // lane d writes p[d][c][n]
        float v = (lane == 0) ? acc.x : (lane == 1) ? acc.y : (lane == 2) ? acc.z : acc.w;
        int n = b * T_ + t;
        p16[(size_t)lane * C_ * N_ + (size_t)c * N_ + n] = (_Float16)(v * (1.f / F_));
    }
}

// ---------------------------------------------------------------------------
// WMMA GEMM for the 1x1 convs: per depth d, Y[o,n] = sum_i W[d][o][i]*P[d][i][n] + bias.
// One wave per 16x16 output tile; K=64 done as two v_wmma_f32_16x16x32_f16 steps.
// Fragment layouts per CDNA5 ISA 7.12.2 (wave32).
// ---------------------------------------------------------------------------
__global__ void gemm_kernel(const float* __restrict__ W,      // [D][C][C] f32
                            const _Float16* __restrict__ Bm,  // [D][C][N] f16
                            const float* __restrict__ bias,   // [D][C]
                            float* __restrict__ Y) {          // [D][C][N] f32
    int wid  = (blockIdx.x * blockDim.x + threadIdx.x) >> 5;  // tile id
    int lane = threadIdx.x & 31;
    int nt = wid % (N_ / 16);
    int mo = (wid / (N_ / 16)) % (C_ / 16);
    int d  = wid / ((N_ / 16) * (C_ / 16));
    int m  = lane & 15;       // A row within tile / B,C column within tile
    int h  = lane >> 4;       // lane half selector

    const float*    Wd = W  + (size_t)d * C_ * C_;
    const _Float16* Bd = Bm + (size_t)d * C_ * N_;

    v8f acc = {};
#pragma unroll
    for (int kb = 0; kb < 2; ++kb) {                  // K = 64 = 2 x 32
        v16h af, bf;
#pragma unroll
        for (int j = 0; j < 16; ++j) {
            // A (16x32, MxK): halves j<8 -> k=h*8+j ; j>=8 -> k=16+h*8+(j-8)
            int ka = (j < 8) ? (h * 8 + j) : (16 + h * 8 + (j - 8));
            af[j] = (_Float16)Wd[(mo * 16 + m) * C_ + kb * 32 + ka];
            // B (32x16, KxN): same k order as A (j<8 -> k=h*8+j ; j>=8 -> k=16+h*8+(j-8)), column = lane%16
            int kbf = (j < 8) ? (h * 8 + j) : (16 + h * 8 + (j - 8));
            bf[j] = Bd[(size_t)(kb * 32 + kbf) * N_ + nt * 16 + m];
        }
        acc = __builtin_amdgcn_wmma_f32_16x16x32_f16(
            /*neg_a=*/false, af, /*neg_b=*/false, bf,
            /*c_mod=*/(short)0, acc, /*reuse_a=*/false, /*reuse_b=*/false);
    }
    // D layout: VGPR r -> M = r + 8*h ; N = lane%16
#pragma unroll
    for (int r = 0; r < 8; ++r) {
        int o = mo * 16 + r + 8 * h;
        Y[(size_t)d * C_ * N_ + (size_t)o * N_ + nt * 16 + m] = acc[r] + bias[d * C_ + o];
    }
}

// ---------------------------------------------------------------------------
// LayerNorm over channels (C=64) + gamma/beta + PReLU, one wave per (d, token).
// Each lane owns channels {lane, lane+32}; wave32 butterfly for mean/var.
// Output as f16 [D][C][N] feeding the second WMMA GEMM.
// ---------------------------------------------------------------------------
__global__ void ln_prelu_kernel(const float* __restrict__ Y,
                                const float* __restrict__ gamma,
                                const float* __restrict__ beta,
                                const float* __restrict__ alpha,
                                _Float16* __restrict__ Yout) {
    int wid  = (blockIdx.x * blockDim.x + threadIdx.x) >> 5;
    int lane = threadIdx.x & 31;
    int n = wid % N_;
    int d = wid / N_;
    const float* Yd = Y + (size_t)d * C_ * N_ + n;
    int c0 = lane, c1 = lane + 32;
    float v0 = Yd[(size_t)c0 * N_];
    float v1 = Yd[(size_t)c1 * N_];
    float s = v0 + v1;
    float q = v0 * v0 + v1 * v1;
#pragma unroll
    for (int off = 16; off > 0; off >>= 1) {
        s += __shfl_xor(s, off);
        q += __shfl_xor(q, off);
    }
    float mu  = s * (1.f / C_);
    float var = q * (1.f / C_) - mu * mu;
    float rs  = rsqrtf(var + EPSV);

    _Float16* Od = Yout + (size_t)d * C_ * N_ + n;
    {
        float yn = (v0 - mu) * rs * gamma[d * C_ + c0] + beta[d * C_ + c0];
        yn = (yn >= 0.f) ? yn : alpha[d * C_ + c0] * yn;
        Od[(size_t)c0 * N_] = (_Float16)yn;
    }
    {
        float yn = (v1 - mu) * rs * gamma[d * C_ + c1] + beta[d * C_ + c1];
        yn = (yn >= 0.f) ? yn : alpha[d * C_ + c1] * yn;
        Od[(size_t)c1 * N_] = (_Float16)yn;
    }
}

// ---------------------------------------------------------------------------
// Softmax over the F' axis, which has size 1 -> exp(v-v)/exp(v-v) (==1, NaN-safe).
// Reorders [D][C][N] -> mask[(b*C+c)*T+t][d] so the final pass can float4-load it.
// ---------------------------------------------------------------------------
__global__ void softmax_mask_kernel(const float* __restrict__ Y2, float* __restrict__ mask) {
    int idx = blockIdx.x * blockDim.x + threadIdx.x;   // over D*C*N
    if (idx >= D_ * C_ * N_) return;
    int n = idx % N_;
    int c = (idx / N_) % C_;
    int d = idx / (C_ * N_);
    float v = Y2[idx];
    float e = __expf(v - v);           // singleton-axis softmax: max == v
    float mval = e / e;                // == 1.0 (NaN propagates like the reference)
    int b = n / T_, t = n % T_;
    mask[(size_t)((b * C_ + c) * T_ + t) * D_ + d] = mval;
}

// ---------------------------------------------------------------------------
// Pass 2: out[b,c,t,f,0] = sum_d x[b,c,t,f,d] * mask[b,c,t,d].
// One wave per (b,c,t) row; float4 loads of x, coalesced 128B stores of out.
// ---------------------------------------------------------------------------
__global__ void apply_mask_kernel(const float* __restrict__ x,
                                  const float* __restrict__ mask,
                                  float* __restrict__ out) {
    int wid  = (blockIdx.x * blockDim.x + threadIdx.x) >> 5;  // row = (b*C+c)*T+t
    int lane = threadIdx.x & 31;
    float4 mk = ((const float4*)mask)[wid];
    const float4* xr = (const float4*)(x + (size_t)wid * (F_ * D_));
    float* orow = out + (size_t)wid * F_;
#pragma unroll
    for (int i = 0; i < F_ / 32; ++i) {
        int f = lane + 32 * i;
        float4 v = xr[f];
        orow[f] = v.x * mk.x + v.y * mk.y + v.z * mk.z + v.w * mk.w;
    }
}

// ---------------------------------------------------------------------------
// Workspace layout (8 MB total):
//   [0,1MB)  p16   f16 [D][C][N]   pooled means
//   [1,3MB)  y1    f32 [D][C][N]   conv1 output
//   [3,4MB)  yln16 f16 [D][C][N]   LN+PReLU output
//   [4,6MB)  y2    f32 [D][C][N]   conv2 output
//   [6,8MB)  mask  f32 [B*C*T][D]  softmax result (==1)
// ---------------------------------------------------------------------------
extern "C" void kernel_launch(void* const* d_in, const int* in_sizes, int n_in,
                              void* d_out, int out_size, void* d_ws, size_t ws_size,
                              hipStream_t stream) {
    const float* x     = (const float*)d_in[0];
    const float* w1    = (const float*)d_in[1];
    const float* b1    = (const float*)d_in[2];
    const float* gamma = (const float*)d_in[3];
    const float* beta  = (const float*)d_in[4];
    const float* alpha = (const float*)d_in[5];
    const float* w2    = (const float*)d_in[6];
    const float* b2    = (const float*)d_in[7];
    float* out = (float*)d_out;

    char* ws = (char*)d_ws;
    _Float16* p16   = (_Float16*)(ws);
    float*    y1    = (float*)(ws + (1u << 20));
    _Float16* yln16 = (_Float16*)(ws + (3u << 20));
    float*    y2    = (float*)(ws + (4u << 20));
    float*    mask  = (float*)(ws + (6u << 20));

    const int rows = B_ * C_ * T_;                 // 131072 (b,c,t) rows
    const int gemm_tiles = D_ * (C_ / 16) * (N_ / 16);  // 2048 wave-tiles

    pool_kernel<<<rows / 8, 256, 0, stream>>>(x, p16);
    gemm_kernel<<<gemm_tiles / 4, 128, 0, stream>>>(w1, p16, b1, y1);
    ln_prelu_kernel<<<(D_ * N_) / 8, 256, 0, stream>>>(y1, gamma, beta, alpha, yln16);
    gemm_kernel<<<gemm_tiles / 4, 128, 0, stream>>>(w2, yln16, b2, y2);
    softmax_mask_kernel<<<(D_ * C_ * N_) / 256, 256, 0, stream>>>(y2, mask);
    apply_mask_kernel<<<rows / 8, 256, 0, stream>>>(x, mask, out);
}
